// ScalableMoE_300647711146
// MI455X (gfx1250) — matrix operand order repaired and checked by reading, not compiled, not run on hardware
//
#include <hip/hip_runtime.h>
#include <hip/hip_bf16.h>

typedef __attribute__((ext_vector_type(16))) __bf16 v16bf;
typedef __attribute__((ext_vector_type(8)))  float  v8f;
typedef __attribute__((ext_vector_type(4)))  int    v4i;

union FragU { v16bf v; v4i q[2]; };

static __device__ __forceinline__ unsigned short f32_to_bf16(float f) {
  unsigned int u = __float_as_uint(f);
  u += 0x7FFFu + ((u >> 16) & 1u);   // round-to-nearest-even
  return (unsigned short)(u >> 16);
}

// CDNA5 async copy: 16B global -> LDS per lane, tracked by ASYNCcnt.
static __device__ __forceinline__ void async_copy16(unsigned lds_off,
                                                    const unsigned short* g) {
  asm volatile("global_load_async_to_lds_b128 %0, %1, off"
               :: "v"(lds_off), "v"(g)
               : "memory");
}
static __device__ __forceinline__ void wait_async0() {
  asm volatile("s_wait_asynccnt 0x0" ::: "memory");
}
static __device__ __forceinline__ unsigned lds_addr32(const void* p) {
  return (unsigned)(size_t)p;   // low 32 bits of generic addr = LDS byte offset
}

// ---------------------------------------------------------------------------
// f32 -> bf16 elementwise convert (x activations), 4 elems/thread
// ---------------------------------------------------------------------------
__global__ __launch_bounds__(256)
void cvt_x_kernel(const float4* __restrict__ x4, unsigned long long* __restrict__ out,
                  long long total4) {
  long long i = (long long)blockIdx.x * 256 + threadIdx.x;
  if (i >= total4) return;
  float4 v = x4[i];
  unsigned long long a = f32_to_bf16(v.x);
  unsigned long long b = f32_to_bf16(v.y);
  unsigned long long c = f32_to_bf16(v.z);
  unsigned long long d = f32_to_bf16(v.w);
  out[i] = a | (b << 16) | (c << 32) | (d << 48);
}

// ---------------------------------------------------------------------------
// [E,R,C] f32  ->  [E,C,R] bf16 (tiled LDS transpose, coalesced both ways)
// ---------------------------------------------------------------------------
__global__ __launch_bounds__(256)
void transpose_cvt_kernel(const float* __restrict__ in, unsigned short* __restrict__ out,
                          int R, int C) {
  __shared__ float t[32][33];
  const size_t plane = (size_t)R * C;
  const float* inE = in + (size_t)blockIdx.z * plane;
  unsigned short* outE = out + (size_t)blockIdx.z * plane;
  const int c0 = blockIdx.x * 32, r0 = blockIdx.y * 32;
  const int tx = threadIdx.x, ty = threadIdx.y;
#pragma unroll
  for (int j = 0; j < 32; j += 8)
    t[ty + j][tx] = inE[(size_t)(r0 + ty + j) * C + (c0 + tx)];
  __syncthreads();
#pragma unroll
  for (int j = 0; j < 32; j += 8)
    outE[(size_t)(c0 + ty + j) * R + (r0 + tx)] = f32_to_bf16(t[tx][ty + j]);
}

// ---------------------------------------------------------------------------
// Router: one wave per token. logits = x[n].rw[e], softmax, top-2, renorm.
// ---------------------------------------------------------------------------
__global__ __launch_bounds__(256)
void router_kernel(const float* __restrict__ x, const float* __restrict__ rw,
                   float* __restrict__ kprob, int* __restrict__ kidx,
                   int Ntok, int H) {
  const int lane = threadIdx.x & 31;
  const int n = blockIdx.x * 8 + (threadIdx.x >> 5);
  if (n >= Ntok) return;
  const float* xr = x + (size_t)n * H;
  float acc[8] = {0.f, 0.f, 0.f, 0.f, 0.f, 0.f, 0.f, 0.f};
  for (int k = lane; k < H; k += 32) {
    float xv = xr[k];
#pragma unroll
    for (int e = 0; e < 8; ++e) acc[e] += xv * rw[e * H + k];
  }
#pragma unroll
  for (int e = 0; e < 8; ++e) {
#pragma unroll
    for (int ofs = 16; ofs > 0; ofs >>= 1) acc[e] += __shfl_xor(acc[e], ofs, 32);
  }
  if (lane == 0) {
    float mx = acc[0];
#pragma unroll
    for (int e = 1; e < 8; ++e) mx = fmaxf(mx, acc[e]);
    float p[8];
#pragma unroll
    for (int e = 0; e < 8; ++e) p[e] = __expf(acc[e] - mx);
    int i0 = 0; float b0 = p[0];
#pragma unroll
    for (int e = 1; e < 8; ++e) if (p[e] > b0) { b0 = p[e]; i0 = e; }
    int i1 = -1; float b1 = -1.f;
#pragma unroll
    for (int e = 0; e < 8; ++e) if (e != i0 && p[e] > b1) { b1 = p[e]; i1 = e; }
    float r = b0 + b1;                 // softmax denominator cancels in renorm
    kprob[2 * n]     = b0 / r;
    kprob[2 * n + 1] = b1 / r;
    kidx[2 * n]     = i0;
    kidx[2 * n + 1] = i1;
  }
}

// ---------------------------------------------------------------------------
// Deterministic capacity dispatch: sequential per-expert rank over P pairs.
// Single 1024-thread block, 16 pairs/thread, LDS Hillis-Steele scan of 8 hists.
// ---------------------------------------------------------------------------
__global__ __launch_bounds__(1024)
void dispatch_kernel(const int* __restrict__ kidx, int* __restrict__ pair_slot,
                     int* __restrict__ row_token, int P, int capacity, int Mpad) {
  __shared__ int hist[1024][8];
  const int tid = threadIdx.x;
  for (int i = tid; i < 8 * Mpad; i += 1024) row_token[i] = -1;

  const int base = tid * 16;
  int e_loc[16];
  int cnt[8] = {0, 0, 0, 0, 0, 0, 0, 0};
#pragma unroll
  for (int j = 0; j < 16; ++j) {
    int p = base + j;
    int e = (p < P) ? kidx[p] : -1;
    e_loc[j] = e;
    if (e >= 0) cnt[e]++;
  }
#pragma unroll
  for (int e = 0; e < 8; ++e) hist[tid][e] = cnt[e];
  __syncthreads();
  for (int ofs = 1; ofs < 1024; ofs <<= 1) {
    int v[8];
    if (tid >= ofs) {
#pragma unroll
      for (int e = 0; e < 8; ++e) v[e] = hist[tid - ofs][e];
    }
    __syncthreads();
    if (tid >= ofs) {
#pragma unroll
      for (int e = 0; e < 8; ++e) hist[tid][e] += v[e];
    }
    __syncthreads();
  }
  int bs[8];
#pragma unroll
  for (int e = 0; e < 8; ++e) bs[e] = (tid > 0) ? hist[tid - 1][e] : 0;

#pragma unroll
  for (int j = 0; j < 16; ++j) {
    int e = e_loc[j];
    if (e < 0) continue;
    int rank = bs[e]++;
    int p = base + j;
    if (rank < capacity) {
      pair_slot[p] = rank;
      row_token[e * Mpad + rank] = p >> 1;   // token id
    } else {
      pair_slot[p] = -1;                     // dropped
    }
  }
}

// ---------------------------------------------------------------------------
// bf16 WMMA grouped GEMM, async-to-LDS pipelined.
//   A   : bf16 rows of length K (GATHER: rows via row_token, -1 => zero page)
//   Bt  : bf16 [E, N, K]  (N-major pre-transposed weights)
//   Cout: SILU ? bf16 [E,Mpad,N] with silu : f32 [E,Mpad,N]
// Block tile 128x256 (8 waves), wave tile 64x64 (16 WMMA / K-step), KB=32,
// double-buffered LDS fed by global_load_async_to_lds_b128 (ASYNCcnt).
// ---------------------------------------------------------------------------
#define TM 128
#define TN 256
#define KB 32
#define LROW 40   // 80B LDS row stride: bank-conflict-free b128 fragment reads

template <bool GATHER, bool SILU>
__global__ __launch_bounds__(256)
void moe_gemm_wmma(const unsigned short* __restrict__ A,
                   const unsigned short* __restrict__ Bt,
                   void* __restrict__ Cout,
                   const int* __restrict__ row_token,
                   const unsigned short* __restrict__ zrow,
                   int Mpad, int N, int K, int lda) {
  __shared__ __align__(16) unsigned short sA[2][TM][LROW];
  __shared__ __align__(16) unsigned short sB[2][TN][LROW];

  const int e   = blockIdx.z;
  const int m0  = blockIdx.y * TM;
  const int n0  = blockIdx.x * TN;
  const int tid = threadIdx.x;

  const unsigned short* BtE = Bt + (size_t)e * N * K;

  // staging ownership: thread -> A rows {r, r+64}, B rows {r, r+64, r+128, r+192}
  const int rowA  = tid >> 2;          // 0..63
  const int kcOff = (tid & 3) * 8;     // bf16 elems within KB slice

  const unsigned short* aP0;
  const unsigned short* aP1;
  if constexpr (GATHER) {
    const int* tok = row_token + (size_t)e * Mpad + m0;
    int t0 = tok[rowA];
    int t1 = tok[rowA + 64];
    aP0 = ((t0 >= 0) ? (A + (size_t)t0 * lda) : zrow) + kcOff;
    aP1 = ((t1 >= 0) ? (A + (size_t)t1 * lda) : zrow) + kcOff;
  } else {
    const unsigned short* AE = A + (size_t)e * Mpad * lda;
    aP0 = AE + (size_t)(m0 + rowA) * lda + kcOff;
    aP1 = AE + (size_t)(m0 + rowA + 64) * lda + kcOff;
  }
  const unsigned short* bP0 = BtE + (size_t)(n0 + rowA) * K + kcOff;
  const unsigned short* bP1 = BtE + (size_t)(n0 + rowA + 64) * K + kcOff;
  const unsigned short* bP2 = BtE + (size_t)(n0 + rowA + 128) * K + kcOff;
  const unsigned short* bP3 = BtE + (size_t)(n0 + rowA + 192) * K + kcOff;

  const unsigned lA0 = lds_addr32(&sA[0][rowA][kcOff]);
  const unsigned lA1 = lds_addr32(&sA[0][rowA + 64][kcOff]);
  const unsigned lB0 = lds_addr32(&sB[0][rowA][kcOff]);
  const unsigned lB1 = lds_addr32(&sB[0][rowA + 64][kcOff]);
  const unsigned lB2 = lds_addr32(&sB[0][rowA + 128][kcOff]);
  const unsigned lB3 = lds_addr32(&sB[0][rowA + 192][kcOff]);
  const unsigned strA = (unsigned)(TM * LROW * 2);   // bytes per A buffer
  const unsigned strB = (unsigned)(TN * LROW * 2);   // bytes per B buffer

  const int lane = tid & 31;
  const int ln   = lane & 15;
  const int lh   = lane >> 4;
  const int wid  = tid >> 5;
  const int wm   = (wid & 1) * 64;     // wave M offset in tile
  const int wn   = (wid >> 1) * 64;    // wave N offset in tile

  v8f acc[4][4] = {};

  auto issue_tile = [&](int nb, int kt) {
    const size_t off = (size_t)kt * KB;
    const unsigned da = nb ? strA : 0u;
    const unsigned db = nb ? strB : 0u;
    async_copy16(lA0 + da, aP0 + off);
    async_copy16(lA1 + da, aP1 + off);
    async_copy16(lB0 + db, bP0 + off);
    async_copy16(lB1 + db, bP1 + off);
    async_copy16(lB2 + db, bP2 + off);
    async_copy16(lB3 + db, bP3 + off);
  };

  issue_tile(0, 0);
  wait_async0();
  __syncthreads();

  const int KT = K / KB;
  int buf = 0;
  for (int kt = 0; kt < KT; ++kt) {
    if (kt + 1 < KT) issue_tile(buf ^ 1, kt + 1);   // writes land in the other buffer
    // ---- compute on buf: preload all 8 fragments, then 16 WMMAs ----
    {
      FragU bfr[4];
#pragma unroll
      for (int ni = 0; ni < 4; ++ni) {
        const unsigned short* p = &sB[buf][wn + ni * 16 + ln][lh * 16];
        bfr[ni].q[0] = *(const v4i*)(p);      // K 0..7  (lanes>=16: 16..23)
        bfr[ni].q[1] = *(const v4i*)(p + 8);  // K 8..15 (lanes>=16: 24..31)
      }
      FragU afr[4];
#pragma unroll
      for (int mi = 0; mi < 4; ++mi) {
        const unsigned short* p = &sA[buf][wm + mi * 16 + ln][lh * 16];
        afr[mi].q[0] = *(const v4i*)(p);      // K 0..7  (lanes>=16: 16..23)
        afr[mi].q[1] = *(const v4i*)(p + 8);  // K 8..15 (lanes>=16: 24..31)
      }
#pragma unroll
      for (int mi = 0; mi < 4; ++mi) {
#pragma unroll
        for (int ni = 0; ni < 4; ++ni) {
          acc[mi][ni] = __builtin_amdgcn_wmma_f32_16x16x32_bf16(
              false, afr[mi].v, false, bfr[ni].v, (short)0, acc[mi][ni], false, false);
        }
      }
    }
    if (kt + 1 < KT) {
      wait_async0();      // this wave's async writes done
      __syncthreads();    // everyone's done; next buffer is fully populated
      buf ^= 1;
    }
  }

  // ---- epilogue: D layout => VGPR v holds M=v (lanes 0-15) / M=v+8 (16-31)
  if constexpr (SILU) {
    unsigned short* CE = (unsigned short*)Cout + (size_t)e * Mpad * N;
#pragma unroll
    for (int mi = 0; mi < 4; ++mi)
#pragma unroll
      for (int ni = 0; ni < 4; ++ni)
#pragma unroll
        for (int v = 0; v < 8; ++v) {
          int r = m0 + wm + mi * 16 + lh * 8 + v;
          int c = n0 + wn + ni * 16 + ln;
          float val = acc[mi][ni][v];
          // silu via fast reciprocal (bf16 output precision)
          float s = val * __builtin_amdgcn_rcpf(1.0f + __expf(-val));
          CE[(size_t)r * N + c] = f32_to_bf16(s);
        }
  } else {
    float* CE = (float*)Cout + (size_t)e * Mpad * N;
#pragma unroll
    for (int mi = 0; mi < 4; ++mi)
#pragma unroll
      for (int ni = 0; ni < 4; ++ni)
#pragma unroll
        for (int v = 0; v < 8; ++v) {
          int r = m0 + wm + mi * 16 + lh * 8 + v;
          int c = n0 + wn + ni * 16 + ln;
          CE[(size_t)r * N + c] = acc[mi][ni][v];
        }
  }
}

// ---------------------------------------------------------------------------
// Combine: out = x + sum_k keep*prob*out_buf[e_k, slot_k]
// ---------------------------------------------------------------------------
__global__ __launch_bounds__(256)
void combine_kernel(const float* __restrict__ x, const float* __restrict__ obuf,
                    const float* __restrict__ kprob, const int* __restrict__ kidx,
                    const int* __restrict__ pslot, float* __restrict__ out,
                    int Ntok, int H, int Mpad) {
  long long idx = (long long)blockIdx.x * blockDim.x + threadIdx.x;
  const int H4 = H >> 2;
  long long total = (long long)Ntok * H4;
  if (idx >= total) return;
  int n  = (int)(idx / H4);
  int h4 = (int)(idx % H4);
  float4 r = ((const float4*)x)[idx];
#pragma unroll
  for (int k = 0; k < 2; ++k) {
    int p = 2 * n + k;
    int s = pslot[p];
    if (s >= 0) {
      int e = kidx[p];
      float w = kprob[p];
      float4 v = ((const float4*)obuf)[((size_t)e * Mpad + s) * H4 + h4];
      r.x += w * v.x; r.y += w * v.y; r.z += w * v.z; r.w += w * v.w;
    }
  }
  ((float4*)out)[idx] = r;
}

// ---------------------------------------------------------------------------
extern "C" void kernel_launch(void* const* d_in, const int* in_sizes, int n_in,
                              void* d_out, int out_size, void* d_ws, size_t ws_size,
                              hipStream_t stream) {
  const float* x  = (const float*)d_in[0];   // [B,T,H]
  const float* rw = (const float*)d_in[1];   // [E,H]
  const float* w1 = (const float*)d_in[2];   // [E,H,F]
  const float* w2 = (const float*)d_in[3];   // [E,F,H]
  float* out = (float*)d_out;

  const int E = 8;
  const int H = in_sizes[1] / E;                                   // 1024
  const long long Ntok = (long long)in_sizes[0] / H;               // 8192
  const int F = (int)((long long)in_sizes[2] / ((long long)E * H));// 4096
  const int P = (int)(Ntok * 2);
  const int capacity = (int)(1.25 * (double)Ntok * 2.0 / (double)E) + 1; // 2561
  const int Mtiles = (capacity + TM - 1) / TM;                     // 21
  const int Mpad = Mtiles * TM;                                    // 2688

  char* wsp = (char*)d_ws;
  auto carve = [&](size_t bytes) {
    void* p = (void*)wsp;
    wsp += (bytes + 255) & ~(size_t)255;
    return p;
  };
  unsigned short* xb  = (unsigned short*)carve((size_t)Ntok * H * 2);
  unsigned short* w1t = (unsigned short*)carve((size_t)E * H * F * 2); // [E,F,H]
  unsigned short* w2t = (unsigned short*)carve((size_t)E * H * F * 2); // [E,H,F]
  unsigned short* hb  = (unsigned short*)carve((size_t)E * Mpad * F * 2);
  float* ob    = (float*)carve((size_t)E * Mpad * H * 4);
  float* kprob = (float*)carve((size_t)P * 4);
  int* kidx    = (int*)carve((size_t)P * 4);
  int* pslot   = (int*)carve((size_t)P * 4);
  int* rowtok  = (int*)carve((size_t)E * Mpad * 4);
  unsigned short* zrow = (unsigned short*)carve(8192);   // zero page for dropped rows

  hipMemsetAsync(zrow, 0, 8192, stream);

  const long long total4 = Ntok * (H / 4);
  cvt_x_kernel<<<dim3((unsigned)((total4 + 255) / 256)), 256, 0, stream>>>(
      (const float4*)x, (unsigned long long*)xb, total4);
  transpose_cvt_kernel<<<dim3(F / 32, H / 32, E), dim3(32, 8), 0, stream>>>(w1, w1t, H, F);
  transpose_cvt_kernel<<<dim3(H / 32, F / 32, E), dim3(32, 8), 0, stream>>>(w2, w2t, F, H);
  router_kernel<<<dim3((unsigned)((Ntok + 7) / 8)), 256, 0, stream>>>(
      x, rw, kprob, kidx, (int)Ntok, H);
  dispatch_kernel<<<1, 1024, 0, stream>>>(kidx, pslot, rowtok, P, capacity, Mpad);

  // GEMM1: h = silu(gather(x) @ w1), bf16 out
  moe_gemm_wmma<true, true><<<dim3(F / TN, Mtiles, E), 256, 0, stream>>>(
      xb, w1t, (void*)hb, rowtok, zrow, Mpad, F, H, H);
  // GEMM2: out_buf = h @ w2, f32 out
  moe_gemm_wmma<false, false><<<dim3(H / TN, Mtiles, E), 256, 0, stream>>>(
      hb, w2t, (void*)ob, nullptr, nullptr, Mpad, H, F, F);

  combine_kernel<<<dim3((unsigned)((total4 + 255) / 256)), 256, 0, stream>>>(
      x, ob, kprob, kidx, pslot, out, (int)Ntok, H, Mpad);
}
